// GWFFN_18932215840902
// MI455X (gfx1250) — matrix operand order repaired and checked by reading, not compiled, not run on hardware
//
#include <hip/hip_runtime.h>

// ---------------------------------------------------------------------------
// GWFFN spiking block for MI455X (gfx1250, wave32, WMMA, async LDS copies).
// x [T=4,N=16,C=128,32x32] f32 ; inner=512 ; groups=8.
// Spikes {0,1} stored f16 (exact); weights pre-converted f32->f16; GEMMs on
// v_wmma_f32_16x16x32_f16 with f32 accumulate; BN/residual fused in epilogue.
// ---------------------------------------------------------------------------

typedef _Float16 v4h  __attribute__((ext_vector_type(4)));
typedef _Float16 v8h  __attribute__((ext_vector_type(8)));
typedef _Float16 v16h __attribute__((ext_vector_type(16)));
typedef float    v8f  __attribute__((ext_vector_type(8)));

#define HW     1024          // 32*32
#define BN_EPS 1e-5f

// ---- CDNA5 async copy: global -> LDS, 16 bytes per lane, ASYNCcnt-tracked --
__device__ __forceinline__ void async_b128(void* lds_dst, const void* gsrc) {
    unsigned lds_off = (unsigned)(size_t)lds_dst;   // low 32 bits = LDS offset
    asm volatile("global_load_async_to_lds_b128 %0, %1, off"
                 :: "v"(lds_off), "v"(gsrc) : "memory");
}
__device__ __forceinline__ void wait_async() {
    asm volatile("s_wait_asynccnt 0x0" ::: "memory");
}

// ---- WMMA fragment loaders (16-bit A 16x32 / B 32x16 VGPR striping) --------
// A: lane L -> m=L%16, half=L/16; two contiguous 8xf16 chunks at k=half*8 and
// k=16+half*8 (ISA 7.12.2).  As is row-major [m][k], stride 32.
__device__ __forceinline__ v16h load_a_frag(const _Float16* As, int lane, int row_base) {
    int m    = row_base + (lane & 15);
    int half = lane >> 4;
    const _Float16* p = As + m * 32;
    v8h lo = *(const v8h*)(p + half * 8);
    v8h hi = *(const v8h*)(p + 16 + half * 8);
    return __builtin_shufflevector(lo, hi, 0,1,2,3,4,5,6,7,8,9,10,11,12,13,14,15);
}
// B: lane L -> n=L%16, half=L/16; two contiguous 8xf16 chunks at k=half*8 and
// k=16+half*8 (same as A).  Bt is [n][k], stride 32.
__device__ __forceinline__ v16h load_b_frag(const _Float16* Bt, int lane, int n_base) {
    int n    = n_base + (lane & 15);
    int half = lane >> 4;
    const _Float16* p = Bt + n * 32;
    v8h lo = *(const v8h*)(p + half * 8);
    v8h hi = *(const v8h*)(p + 16 + half * 8);
    return __builtin_shufflevector(lo, hi, 0,1,2,3,4,5,6,7,8,9,10,11,12,13,14,15);
}

__device__ __forceinline__ v8f wmma_f16(v16h a, v16h b, v8f c) {
    return __builtin_amdgcn_wmma_f32_16x16x32_f16(false, a, false, b, (short)0, c, false, false);
}

// ---- weight convert f32 -> f16 (vectorized) --------------------------------
__global__ void cvt_f32_f16(const float* __restrict__ src, _Float16* __restrict__ dst, int n4) {
    int i = blockIdx.x * blockDim.x + threadIdx.x;
    if (i >= n4) return;
    float4 v = ((const float4*)src)[i];
    v4h o = {(_Float16)v.x, (_Float16)v.y, (_Float16)v.z, (_Float16)v.w};
    ((v4h*)dst)[i] = o;
}

// ---- LIF: v += (x-v)/2 ; spike=(v>=1) ; hard reset.  4 elems/thread --------
__global__ void lif_kernel(const float* __restrict__ x, _Float16* __restrict__ s, int C) {
    int i = blockIdx.x * blockDim.x + threadIdx.x;
    int q = C * 4096;                      // float4 per time slice (16*C*1024/4)
    if (i >= q) return;
    float4 v = {0.f, 0.f, 0.f, 0.f};
#pragma unroll
    for (int t = 0; t < 4; ++t) {
        float4 xt = ((const float4*)x)[t * q + i];
        v.x = 0.5f * (v.x + xt.x);  v.y = 0.5f * (v.y + xt.y);
        v.z = 0.5f * (v.z + xt.z);  v.w = 0.5f * (v.w + xt.w);
        float s0 = (v.x >= 1.f) ? 1.f : 0.f,  s1 = (v.y >= 1.f) ? 1.f : 0.f;
        float s2 = (v.z >= 1.f) ? 1.f : 0.f,  s3 = (v.w >= 1.f) ? 1.f : 0.f;
        v4h o = {(_Float16)s0, (_Float16)s1, (_Float16)s2, (_Float16)s3};
        ((v4h*)s)[t * q + i] = o;
        v.x = (s0 > 0.f) ? 0.f : v.x;  v.y = (s1 > 0.f) ? 0.f : v.y;
        v.z = (s2 > 0.f) ? 0.f : v.z;  v.w = (s3 > 0.f) ? 0.f : v.w;
    }
}

// ---- Dense 1x1 conv GEMM: D[M,L]=W16[M,K]*S[K,L]; BN(+resid) epilogue ------
// 256 threads = 8 waves (4 in M x 2 in N); block tile 128(M) x 128(N);
// wave tile 32x64 = 8 WMMAs per K-step; K-step 32; double-buffered LDS.
__global__ __launch_bounds__(256) void gemm1x1_wmma(
    const _Float16* __restrict__ S,     // spikes  [TN, K, HW] f16
    const _Float16* __restrict__ W16,   // weights [M, K] f16
    const float* __restrict__ gamma, const float* __restrict__ beta,
    const float* __restrict__ mean,  const float* __restrict__ var,
    const float* __restrict__ resid,    // [TN, M, HW] f32 or nullptr
    float*       __restrict__ out,      // [TN, M, HW] f32
    int M, int K)
{
    __shared__ __align__(32) _Float16 As[2][128 * 32];
    __shared__ __align__(32) _Float16 Bt[2][128 * 32];

    const int tid  = threadIdx.x;
    const int lane = tid & 31;
    const int wave = tid >> 5;
    const int wm   = wave & 3;          // 4 waves along M (32 rows each)
    const int wn   = wave >> 2;         // 2 waves along N (64 cols each)
    const int m0   = blockIdx.y * 128;
    const int n0   = blockIdx.x * 128;
    const int tn   = n0 >> 10;          // 128 | 1024 -> tile within one image
    const int hw0  = n0 & 1023;
    const int nk   = K >> 5;

    auto fill = [&](int kt, int buf) {
        const int k0 = kt << 5;
        // A tile 128x32: 512 b128 chunks, 2 per thread, async copy
#pragma unroll
        for (int i = 0; i < 2; ++i) {
            int c = i * 256 + tid;
            int row = c >> 2, col = (c & 3) * 8;
            async_b128(&As[buf][row * 32 + col], &W16[(m0 + row) * K + k0 + col]);
        }
        // B tile 32x128 spikes: 2 x (v8h load + transposed b16 stores) / thread
#pragma unroll
        for (int i = 0; i < 2; ++i) {
            int c = i * 256 + tid;
            int k = c >> 4, n8 = (c & 15) * 8;
            v8h sv = *(const v8h*)&S[tn * (K * HW) + (k0 + k) * HW + hw0 + n8];
#pragma unroll
            for (int e = 0; e < 8; ++e)
                Bt[buf][(n8 + e) * 32 + k] = sv[e];
        }
    };

    v8f acc[2][4] = {};
    fill(0, 0);
    for (int kt = 0; kt < nk; ++kt) {
        const int cur = kt & 1;
        wait_async();
        __syncthreads();                    // buf `cur` ready for everyone
        if (kt + 1 < nk) fill(kt + 1, cur ^ 1);
        // Preload all fragments, then run the 8-WMMA chain under one wait.
        v16h a0 = load_a_frag(As[cur], lane, wm * 32);
        v16h a1 = load_a_frag(As[cur], lane, wm * 32 + 16);
        v16h b[4];
#pragma unroll
        for (int nt = 0; nt < 4; ++nt)
            b[nt] = load_b_frag(Bt[cur], lane, wn * 64 + nt * 16);
#pragma unroll
        for (int nt = 0; nt < 4; ++nt) acc[0][nt] = wmma_f16(a0, b[nt], acc[0][nt]);
#pragma unroll
        for (int nt = 0; nt < 4; ++nt) acc[1][nt] = wmma_f16(a1, b[nt], acc[1][nt]);
    }

    // Epilogue: BN + optional residual.  D layout: VGPR j -> M = 8*(lane/16)+j.
    const int half = lane >> 4;
#pragma unroll
    for (int mt = 0; mt < 2; ++mt) {
#pragma unroll
        for (int j = 0; j < 8; ++j) {
            int m = m0 + wm * 32 + mt * 16 + half * 8 + j;
            float sc = gamma[m] * rsqrtf(var[m] + BN_EPS);
            float sh = beta[m] - mean[m] * sc;
#pragma unroll
            for (int nt = 0; nt < 4; ++nt) {
                int col = n0 + wn * 64 + nt * 16 + (lane & 15);
                int hw  = col & 1023;
                int o   = tn * (M * HW) + m * HW + hw;
                float v = acc[mt][nt][j] * sc + sh;
                if (resid) v += resid[o];
                out[o] = v;
            }
        }
    }
}

// ---- Grouped 3x3 conv (8 groups, 64->64) as implicit GEMM ------------------
// Per group: M=64, K=576 (ci*9+kh*3+kw, contiguous weights).  Block: 8 waves
// (4 in M x 2 in N); tile 64(M) x 256(N); wave tile 16x128 = 8 WMMAs/step.
// Input patch (64ci x 10rows x 32w = 40 KB) staged in LDS once per block.
__global__ __launch_bounds__(256) void conv3x3g_wmma(
    const _Float16* __restrict__ S,     // spikes [TN, 512, HW] f16
    const _Float16* __restrict__ Wc16,  // weights [512, 576] f16
    const float* __restrict__ gamma, const float* __restrict__ beta,
    const float* __restrict__ mean,  const float* __restrict__ var,
    float* __restrict__ hbuf)           // [TN, 512, HW] f32  in-place (+resid)
{
    __shared__ __align__(32) _Float16 As[2][64 * 32];       //  4 KB each
    __shared__ __align__(32) _Float16 Bt[2][256 * 32];      // 16 KB each
    __shared__ __align__(32) _Float16 Patch[64 * 10 * 32];  // 40 KB

    const int tid  = threadIdx.x;
    const int lane = tid & 31;
    const int wave = tid >> 5;
    const int wm   = wave & 3;          // 4 waves along M (16 rows each)
    const int wn   = wave >> 2;         // 2 waves along N (128 cols each)
    const int g    = blockIdx.y;
    const int n0   = blockIdx.x * 256;
    const int tn   = n0 >> 10;
    const int hw0  = n0 & 1023;
    const int h0   = hw0 >> 5;          // first of 8 output rows in this tile

    // Stage input patch: rows h0-1 .. h0+8 for all 64 group channels.
    // 2560 b128 chunks, 10 per thread; OOB rows zero-filled.
#pragma unroll
    for (int i = 0; i < 10; ++i) {
        int c   = i * 256 + tid;         // 0..2559
        int w8  = (c & 3) * 8;
        int row = c >> 2;                // 0..639 = ci*10 + hr
        int hr  = row % 10;
        int ci  = row / 10;
        int hs  = h0 - 1 + hr;
        _Float16* dst = &Patch[(ci * 10 + hr) * 32 + w8];
        if ((unsigned)hs < 32u) {
            async_b128(dst, &S[tn * (512 * HW) + (g * 64 + ci) * HW + hs * 32 + w8]);
        } else {
            v8h z = {};
            *(v8h*)dst = z;
        }
    }

    auto fill = [&](int kt, int buf) {
        const int k0 = kt << 5;
        // A tile 64x32: 256 b128 chunks, 1 per thread, async copy
        {
            int row = tid >> 2, col = (tid & 3) * 8;
            async_b128(&As[buf][row * 32 + col], &Wc16[(g * 64 + row) * 576 + k0 + col]);
        }
        // B tile 32x256 im2col from LDS patch: 4 slots of (1 k, 8 n) per thread
#pragma unroll
        for (int i = 0; i < 4; ++i) {
            int slot = i * 256 + tid;    // 0..1023
            int k  = slot >> 5;          // 0..31
            int n8 = (slot & 31) * 8;    // 0..248
            int kk = k0 + k;
            int ci = kk / 9;
            int rr = kk - ci * 9;
            int kh = rr / 3;
            int kw = rr - kh * 3;
            int hr = (n8 >> 5) + kh;     // patch row = (h-h0) + kh (with -1 folded)
            int wb = n8 & 31;
            const _Float16* prow = &Patch[(ci * 10 + hr) * 32];
            _Float16 vals[8];
#pragma unroll
            for (int e = 0; e < 8; ++e) {
                int ws = wb + e + kw - 1;
                vals[e] = ((unsigned)ws < 32u) ? prow[ws] : (_Float16)0.f;
            }
#pragma unroll
            for (int e = 0; e < 8; ++e)
                Bt[buf][(n8 + e) * 32 + k] = vals[e];
        }
    };

    wait_async();
    __syncthreads();                      // patch ready
    v8f acc[8] = {};
    fill(0, 0);
    for (int kt = 0; kt < 18; ++kt) {     // 576 / 32
        const int cur = kt & 1;
        wait_async();
        __syncthreads();
        if (kt + 1 < 18) fill(kt + 1, cur ^ 1);
        v16h a = load_a_frag(As[cur], lane, wm * 16);
        // two groups of (preload 4 b-frags -> 4 WMMAs) to bound live registers
#pragma unroll
        for (int gb = 0; gb < 2; ++gb) {
            v16h b[4];
#pragma unroll
            for (int q = 0; q < 4; ++q)
                b[q] = load_b_frag(Bt[cur], lane, wn * 128 + (gb * 4 + q) * 16);
#pragma unroll
            for (int q = 0; q < 4; ++q)
                acc[gb * 4 + q] = wmma_f16(a, b[q], acc[gb * 4 + q]);
        }
    }

    // Epilogue: BN + residual, in place on hbuf (disjoint elements per block)
    const int half = lane >> 4;
#pragma unroll
    for (int j = 0; j < 8; ++j) {
        int m = g * 64 + wm * 16 + half * 8 + j;
        float sc = gamma[m] * rsqrtf(var[m] + BN_EPS);
        float sh = beta[m] - mean[m] * sc;
#pragma unroll
        for (int nt = 0; nt < 8; ++nt) {
            int col = n0 + wn * 128 + nt * 16 + (lane & 15);
            int hw  = col & 1023;
            int o   = tn * (512 * HW) + m * HW + hw;
            hbuf[o] = acc[nt][j] * sc + sh + hbuf[o];
        }
    }
}

// ---------------------------------------------------------------------------
extern "C" void kernel_launch(void* const* d_in, const int* in_sizes, int n_in,
                              void* d_out, int out_size, void* d_ws, size_t ws_size,
                              hipStream_t stream) {
    const float* x      = (const float*)d_in[0];
    const float* w_up   = (const float*)d_in[1];
    const float* g_up   = (const float*)d_in[2];
    const float* b_up   = (const float*)d_in[3];
    const float* m_up   = (const float*)d_in[4];
    const float* v_up   = (const float*)d_in[5];
    const float* w_conv = (const float*)d_in[6];
    const float* g_cv   = (const float*)d_in[7];
    const float* b_cv   = (const float*)d_in[8];
    const float* m_cv   = (const float*)d_in[9];
    const float* v_cv   = (const float*)d_in[10];
    const float* w_dn   = (const float*)d_in[11];
    const float* g_dn   = (const float*)d_in[12];
    const float* b_dn   = (const float*)d_in[13];
    const float* m_dn   = (const float*)d_in[14];
    const float* v_dn   = (const float*)d_in[15];
    float* out = (float*)d_out;

    // Workspace carving (~214 MiB):
    char* ws = (char*)d_ws;
    _Float16* s1   = (_Float16*)ws;                                   // 16 MiB
    float*    hbuf = (float*)(ws + (size_t)16777216);                 // 128 MiB
    _Float16* s2   = (_Float16*)(ws + (size_t)150994944);             // 64 MiB
    _Float16* w16u = (_Float16*)(ws + (size_t)218103808);             // 128 KiB
    _Float16* w16c = (_Float16*)(ws + (size_t)218234880);             // 5.625 MiB
    _Float16* w16d = (_Float16*)(ws + (size_t)224133120);             // 128 KiB

    // 0) weights f32 -> f16 (once per call)
    cvt_f32_f16<<<64,   256, 0, stream>>>(w_up,   w16u, 16384);
    cvt_f32_f16<<<2880, 256, 0, stream>>>(w_conv, w16c, 737280);
    cvt_f32_f16<<<64,   256, 0, stream>>>(w_dn,   w16d, 16384);

    // 1) s1 = LIF(x)
    lif_kernel<<<2048, 256, 0, stream>>>(x, s1, 128);
    // 2) h1 = BN(conv1x1_up(s1))          M=512, K=128
    gemm1x1_wmma<<<dim3(512, 4), 256, 0, stream>>>(
        s1, w16u, g_up, b_up, m_up, v_up, nullptr, hbuf, 512, 128);
    // 3) s2 = LIF(h1)
    lif_kernel<<<8192, 256, 0, stream>>>(hbuf, s2, 512);
    // 4) h2 = BN(gconv3x3(s2)) + h1       (in place on hbuf)
    conv3x3g_wmma<<<dim3(256, 8), 256, 0, stream>>>(
        s2, w16c, g_cv, b_cv, m_cv, v_cv, hbuf);
    // 5) s3 = LIF(h2)                     (reuse s2 buffer)
    lif_kernel<<<8192, 256, 0, stream>>>(hbuf, s2, 512);
    // 6) out = BN(conv1x1_down(s3)) + x   M=128, K=512
    gemm1x1_wmma<<<dim3(512, 1), 256, 0, stream>>>(
        s2, w16d, g_dn, b_dn, m_dn, v_dn, x, out, 128, 512);

    (void)in_sizes; (void)n_in; (void)out_size; (void)ws_size;
}
